// XAttn_1889785610810
// MI455X (gfx1250) — matrix operand order repaired and checked by reading, not compiled, not run on hardware
//
#include <hip/hip_runtime.h>
#include <math.h>

// Problem constants (from the reference)
#define N_NODES 10000
#define NPAD    10016          // node count padded to a multiple of 32 (K-dim of k_mp)
#define HID     768
#define SEQ_LEN 16
#define LN_EPS  1e-5f
#define DEG_SPLIT 8            // i-range split for k_deg
#define DEG_CHUNK (N_NODES / DEG_SPLIT)   // 1250

typedef __bf16 bf16;
typedef __attribute__((ext_vector_type(16))) __bf16 v16bf;
typedef __attribute__((ext_vector_type(8)))  __bf16 v8bf;
typedef __attribute__((ext_vector_type(2)))  __bf16 v2bf;
typedef __attribute__((ext_vector_type(8)))  float  v8f;

__device__ __forceinline__ bf16 f2bf(float f) { return (bf16)f; }

__device__ __forceinline__ float gelu_exact(float x) {
    // 0.5*x*(1+erf(x/sqrt(2))) -- exact GELU as in the reference
    return 0.5f * x * (1.0f + erff(x * 0.7071067811865476f));
}

// ---- WMMA fragment builders (16x16x32 bf16, contiguous K order used by both A and B) ----
// A (16x32, MxK): lanes 0-15 hold M=lane, K in {0..15}; lanes 16-31 hold
// M=lane-16, K in {16..31}; each VGPR packs (K, K+1).
// Reads from an LDS tile stored [m][k] -> compiler emits 2x ds_load_b128.
__device__ __forceinline__ v16bf load_a_lds(const bf16* lds, int r, int half) {
    v16bf a;
#pragma unroll
    for (int v = 0; v < 8; ++v) {
        int k = 2 * v + 16 * half;
        a[2 * v]     = lds[r * 32 + k];
        a[2 * v + 1] = lds[r * 32 + k + 1];
    }
    return a;
}

// B (32x16, KxN): per lane, b[0..15] = K = 16*half .. 16*half+15 for column
// N = lane&15. With the source stored TRANSPOSED (N-major, ldT = K-stride),
// that is one contiguous 32-byte run -> 2x global_load_b128.
__device__ __forceinline__ v16bf load_b_T(const bf16* __restrict__ srcT, int ldT,
                                          int kBase, int col, int half) {
    return *(const v16bf*)(srcT + (size_t)col * ldT + kBase + 16 * half);
}

// ---------------------------------------------------------------------------
// Prep: transpose + convert W, W0 (f32 HxH) to bf16 N-major copies.
// ---------------------------------------------------------------------------
__global__ void k_prep_w(const float* __restrict__ W, const float* __restrict__ W0,
                         bf16* __restrict__ WT, bf16* __restrict__ W0T) {
    int id = blockIdx.x * blockDim.x + threadIdx.x;
    if (id >= HID * HID) return;
    int c = id % HID, k = id / HID;          // coalesced reads over c
    WT [(size_t)c * HID + k] = f2bf(W [(size_t)k * HID + c]);
    W0T[(size_t)c * HID + k] = f2bf(W0[(size_t)k * HID + c]);
}

// Zero the K-tail pad of hT (so k_mp's B loads need no guards) and zero deg
// (accumulated with atomics by k_deg).
__global__ void k_zeropad(bf16* __restrict__ hT, float* __restrict__ deg) {
    int id = blockIdx.x * blockDim.x + threadIdx.x;
    if (id < N_NODES) deg[id] = 0.0f;
    const int PAD = NPAD - N_NODES;          // 16
    if (id < HID * PAD) {
        int c = id / PAD, i = N_NODES + (id % PAD);
        hT[(size_t)c * NPAD + i] = (bf16)0.0f;
    }
}

// ---------------------------------------------------------------------------
// Kernel 1: h = semantics[:,0] @ W, stored TRANSPOSED bf16: hT[c*NPAD + n]
// grid(625), block 256 (8 waves); wave w computes c-tiles w*6..w*6+5.
// ---------------------------------------------------------------------------
__global__ void k_gemm_h(const float* __restrict__ sem, const bf16* __restrict__ WT,
                         bf16* __restrict__ hT) {
    __shared__ bf16 aTile[16 * 32];
    const int tid  = threadIdx.x;
    const int lane = tid & 31;
    const int wave = tid >> 5;
    const int r = lane & 15, half = lane >> 4;
    const int mBase = blockIdx.x * 16;

    // Staging coords: thread t owns elements (m = t>>4, kk = (t&15)*2, +1):
    // one float2 load + cvt_pk + one ds_store_b32.
    const int sm = tid >> 4;
    const int sk = (tid & 15) * 2;
    const float* srow = sem + (size_t)(mBase + sm) * (SEQ_LEN * HID);

    v8f acc[6];
    v8f zero = {};
#pragma unroll
    for (int t = 0; t < 6; ++t) acc[t] = zero;

    for (int kt = 0; kt < HID / 32; ++kt) {
        const int kBase = kt * 32;
        __syncthreads();
        {
            float2 v = *(const float2*)(srow + kBase + sk);
            v2bf o; o[0] = f2bf(v.x); o[1] = f2bf(v.y);
            *(v2bf*)(aTile + sm * 32 + sk) = o;
        }
        __syncthreads();
        v16bf a = load_a_lds(aTile, r, half);
#pragma unroll
        for (int t = 0; t < 6; ++t) {
            const int col = (wave * 6 + t) * 16 + r;
            v16bf b = load_b_T(WT, HID, kBase, col, half);
            acc[t] = __builtin_amdgcn_wmma_f32_16x16x32_bf16(
                false, a, false, b, (short)0, acc[t], false, false);
        }
    }
    // D layout: acc[i] -> (row = mBase + i + 8*half, col). Transposed store:
    // 8 consecutive n-positions per lane -> one packed 16B store.
#pragma unroll
    for (int t = 0; t < 6; ++t) {
        const int col = (wave * 6 + t) * 16 + r;
        v8bf o;
#pragma unroll
        for (int i = 0; i < 8; ++i) o[i] = f2bf(acc[t][i]);
        *(v8bf*)(hT + (size_t)col * NPAD + mBase + 8 * half) = o;
    }
}

// ---------------------------------------------------------------------------
// Kernel 2: deg[j] += sum over an i-chunk of (adj[i,j] != 0).
// grid (40, DEG_SPLIT); float atomics are exact for integer counts.
// ---------------------------------------------------------------------------
__global__ void k_deg(const float* __restrict__ adj, float* __restrict__ deg) {
    int j = blockIdx.x * blockDim.x + threadIdx.x;
    if (j >= N_NODES) return;
    int i0 = blockIdx.y * DEG_CHUNK;
    float s = 0.0f;
    for (int i = i0; i < i0 + DEG_CHUNK; ++i)
        s += (adj[(size_t)i * N_NODES + j] != 0.0f) ? 1.0f : 0.0f;
    atomicAdd(&deg[j], s);
}

// ---------------------------------------------------------------------------
// Kernel 3: x1 = gelu( (mask^T @ h) / max(deg,1) ), stored transposed bf16.
// Dense bf16 WMMA over the 0/1 mask (exact in bf16). One block per 16-row
// j-tile owns all 768 output cols -> adj streamed exactly once (400 MB).
// Software-pipelined: next adj tile is loaded to registers under the WMMAs.
// ---------------------------------------------------------------------------
__global__ void k_mp(const float* __restrict__ adj, const bf16* __restrict__ hT,
                     const float* __restrict__ deg, bf16* __restrict__ x1T) {
    __shared__ bf16 aTile[16 * 32];   // [m = j-offset][kk = i-offset], A = mask^T
    const int tid  = threadIdx.x;
    const int lane = tid & 31;
    const int wave = tid >> 5;
    const int r = lane & 15, half = lane >> 4;
    const int jBase = blockIdx.x * 16;

    // Staging coords: thread t owns (m = t&15, kk = t>>4) and (m, kk+16).
    const int sm = tid & 15;
    const int sk = tid >> 4;
    const float* colbase = adj + jBase + sm;

    v8f acc[6];
    v8f zero = {};
#pragma unroll
    for (int t = 0; t < 6; ++t) acc[t] = zero;

    // Prologue: tile 0 (rows sk, sk+16 < N_NODES always).
    float f0 = colbase[(size_t)sk * N_NODES];
    float f1 = colbase[(size_t)(sk + 16) * N_NODES];

    const int KT = NPAD / 32;   // 313; hT pad is zeroed
    for (int kt = 0; kt < KT; ++kt) {
        const int iBase = kt * 32;
        __syncthreads();    // previous iteration's A reads complete
        aTile[sm * 32 + sk]      = (f0 != 0.0f) ? (bf16)1.0f : (bf16)0.0f;
        aTile[sm * 32 + sk + 16] = (f1 != 0.0f) ? (bf16)1.0f : (bf16)0.0f;
        // Issue next tile's loads now; the wait lands at next iteration's store.
        {
            int n0 = iBase + 32 + sk;
            int n1 = n0 + 16;
            int c0 = n0 < N_NODES ? n0 : N_NODES - 1;   // clamp: branch-free guard
            int c1 = n1 < N_NODES ? n1 : N_NODES - 1;
            float v0 = colbase[(size_t)c0 * N_NODES];
            float v1 = colbase[(size_t)c1 * N_NODES];
            f0 = (n0 < N_NODES) ? v0 : 0.0f;
            f1 = (n1 < N_NODES) ? v1 : 0.0f;
        }
        __syncthreads();
        v16bf a = load_a_lds(aTile, r, half);
#pragma unroll
        for (int t = 0; t < 6; ++t) {
            const int col = (wave * 6 + t) * 16 + r;
            v16bf b = load_b_T(hT, NPAD, iBase, col, half);
            acc[t] = __builtin_amdgcn_wmma_f32_16x16x32_bf16(
                false, a, false, b, (short)0, acc[t], false, false);
        }
    }
#pragma unroll
    for (int t = 0; t < 6; ++t) {
        const int col = (wave * 6 + t) * 16 + r;
        v8bf o;
#pragma unroll
        for (int i = 0; i < 8; ++i) {
            int row = jBase + 8 * half + i;
            float s = acc[t][i] / fmaxf(deg[row], 1.0f);
            o[i] = f2bf(gelu_exact(s));
        }
        *(v8bf*)(x1T + (size_t)col * NPAD + jBase + 8 * half) = o;
    }
}

// ---------------------------------------------------------------------------
// Kernel 4: y = gelu(x1 @ W0); LayerNorm(y) @ W1 -> out
// One block per 16-row tile; full 16x768 row-block of y kept in LDS, then
// wave32 shuffle reductions do LN + final dot.
// ---------------------------------------------------------------------------
#define LDY 776   // 768 + 8 pad to dodge bank conflicts
__global__ void k_head(const bf16* __restrict__ x1T, const bf16* __restrict__ W0T,
                       const float* __restrict__ ln_g, const float* __restrict__ ln_b,
                       const float* __restrict__ W1, float* __restrict__ out) {
    __shared__ bf16  aTile[16 * 32];
    __shared__ float yTile[16 * LDY];
    const int tid  = threadIdx.x;
    const int lane = tid & 31;
    const int wave = tid >> 5;
    const int r = lane & 15, half = lane >> 4;
    const int mBase = blockIdx.x * 16;

    // Staging coords: thread t owns (kk = t>>3, m = (t&7)*2, +1):
    // one b32 load of two bf16 from x1T (contiguous in n) + two ds_store_b16.
    const int sk = tid >> 3;
    const int sm = (tid & 7) * 2;

    v8f acc[6];
    v8f zero = {};
#pragma unroll
    for (int t = 0; t < 6; ++t) acc[t] = zero;

    for (int kt = 0; kt < HID / 32; ++kt) {
        const int kBase = kt * 32;
        __syncthreads();
        {
            v2bf v = *(const v2bf*)(x1T + (size_t)(kBase + sk) * NPAD + mBase + sm);
            aTile[sm * 32 + sk]       = v[0];
            aTile[(sm + 1) * 32 + sk] = v[1];
        }
        __syncthreads();
        v16bf a = load_a_lds(aTile, r, half);
#pragma unroll
        for (int t = 0; t < 6; ++t) {
            const int col = (wave * 6 + t) * 16 + r;
            v16bf b = load_b_T(W0T, HID, kBase, col, half);
            acc[t] = __builtin_amdgcn_wmma_f32_16x16x32_bf16(
                false, a, false, b, (short)0, acc[t], false, false);
        }
    }
    // gelu -> LDS row block
#pragma unroll
    for (int t = 0; t < 6; ++t) {
        const int col = (wave * 6 + t) * 16 + r;
#pragma unroll
        for (int i = 0; i < 8; ++i) {
            int row = i + 8 * half;
            yTile[row * LDY + col] = gelu_exact(acc[t][i]);
        }
    }
    __syncthreads();

    // LayerNorm + dot(W1): wave w handles rows 2w, 2w+1 (wave32 shuffles)
#pragma unroll
    for (int rr = 0; rr < 2; ++rr) {
        int row = wave * 2 + rr;
        float s = 0.0f, sq = 0.0f;
        for (int c = lane; c < HID; c += 32) {
            float v = yTile[row * LDY + c];
            s += v; sq += v * v;
        }
#pragma unroll
        for (int off = 16; off > 0; off >>= 1) {
            s  += __shfl_xor(s, off, 32);
            sq += __shfl_xor(sq, off, 32);
        }
        float mu  = s * (1.0f / HID);
        float var = sq * (1.0f / HID) - mu * mu;
        float inv = rsqrtf(var + LN_EPS);
        float d = 0.0f;
        for (int c = lane; c < HID; c += 32) {
            float v = (yTile[row * LDY + c] - mu) * inv * ln_g[c] + ln_b[c];
            d += v * W1[c];
        }
#pragma unroll
        for (int off = 16; off > 0; off >>= 1) d += __shfl_xor(d, off, 32);
        if (lane == 0) out[mBase + row] = d;
    }
}

// ---------------------------------------------------------------------------
extern "C" void kernel_launch(void* const* d_in, const int* in_sizes, int n_in,
                              void* d_out, int out_size, void* d_ws, size_t ws_size,
                              hipStream_t stream) {
    const float* adj  = (const float*)d_in[0];
    const float* sem  = (const float*)d_in[1];
    // d_in[2] attention_masks: unused by the reference
    const float* W    = (const float*)d_in[3];
    const float* W0   = (const float*)d_in[4];
    const float* ln_g = (const float*)d_in[5];
    const float* ln_b = (const float*)d_in[6];
    const float* W1   = (const float*)d_in[7];
    float* out = (float*)d_out;

    char* ws = (char*)d_ws;
    const size_t szT = (size_t)HID * NPAD * sizeof(bf16);    // 15,384,576 B
    const size_t szW = (size_t)HID * HID * sizeof(bf16);     //  1,179,648 B
    bf16*  hT   = (bf16*)(ws);
    bf16*  x1T  = (bf16*)(ws + szT);
    bf16*  WT   = (bf16*)(ws + 2 * szT);
    bf16*  W0T  = (bf16*)(ws + 2 * szT + szW);
    float* deg  = (float*)(ws + 2 * szT + 2 * szW);          // total ~33.2 MB

    dim3 blk(256);
    k_prep_w <<<dim3((HID * HID + 255) / 256), blk, 0, stream>>>(W, W0, WT, W0T);
    k_zeropad<<<dim3((HID * (NPAD - N_NODES) + 255) / 256), blk, 0, stream>>>(hT, deg);
    k_deg    <<<dim3((N_NODES + 255) / 256, DEG_SPLIT), blk, 0, stream>>>(adj, deg);
    k_gemm_h <<<dim3(N_NODES / 16), blk, 0, stream>>>(sem, WT, hT);
    k_mp     <<<dim3(N_NODES / 16), blk, 0, stream>>>(adj, hT, deg, x1T);
    k_head   <<<dim3(N_NODES / 16), blk, 0, stream>>>(x1T, W0T, ln_g, ln_b, W1, out);
}
